// VectorQuantizer_53781580480539
// MI455X (gfx1250) — compile-verified, matrix operand order checked by reading, not run on hardware
//
#include <hip/hip_runtime.h>
#include <hip/hip_bf16.h>
#include <math.h>

// ---------------- problem constants ----------------
#define VQ_D   256
#define VQ_K   1024
#define VQ_N   65536          // 64 * 32 * 32 rows
#define VQ_B   64
#define VQ_HW  1024           // 32*32
#define BETA_F 0.25f

// d_out tuple layout: loss(1) | out(16777216) | perplexity(1) | encodings(67108864)
#define OUT_OFF  1
#define PERP_OFF (1 + VQ_B * VQ_D * VQ_HW)          // 16777217
#define ENC_OFF  (PERP_OFF + 1)                     // 16777218

// ---------------- WMMA types ----------------
typedef __attribute__((ext_vector_type(16))) __bf16 v16bf;
typedef __attribute__((ext_vector_type(8)))  float  v8f;

union Frag16 {
    v16bf v;
    unsigned int u[8];
};

__device__ __forceinline__ unsigned short f32_to_bf16_rne(float f) {
    unsigned int u = __float_as_uint(f);
    u = (u + 0x7FFFu + ((u >> 16) & 1u)) >> 16;
    return (unsigned short)u;
}

// ---------------------------------------------------------------------------
// Kernel 0: codebook -> bf16 (row major [K][D]) + ||e_k||^2 ; zero hist/sumsq
// grid: K blocks x D threads
// ---------------------------------------------------------------------------
__global__ void vq_prep(const float* __restrict__ emb,
                        unsigned short* __restrict__ e_bf,
                        float* __restrict__ enorm,
                        int* __restrict__ hist,
                        float* __restrict__ sumsq) {
    const int k = blockIdx.x;
    const int t = threadIdx.x;
    float e = emb[(size_t)k * VQ_D + t];
    e_bf[(size_t)k * VQ_D + t] = f32_to_bf16_rne(e);

    __shared__ float red[VQ_D];
    red[t] = e * e;
    __syncthreads();
    for (int s = VQ_D / 2; s > 0; s >>= 1) {
        if (t < s) red[t] += red[t + s];
        __syncthreads();
    }
    if (t == 0) {
        enorm[k] = red[0];
        hist[k] = 0;
        if (k == 0) *sumsq = 0.0f;
    }
}

// ---------------------------------------------------------------------------
// Kernel 1: distance + argmin via v_wmma_f32_16x16x32_bf16
//   dist(n,k) = ||e_k||^2 - 2 * x_n . e_k     (||x||^2 dropped: constant per n)
// grid: N/128 blocks x 256 threads (8 waves, 16 rows each)
// ---------------------------------------------------------------------------
#define LDS_ROW 272   // ushorts per LDS row (256 + pad); 544B row = 16B aligned

__global__ __launch_bounds__(256, 1)
void vq_dist_argmin(const float* __restrict__ inputs,        // [B][D][HW]
                    const unsigned short* __restrict__ e_bf, // [K][D] bf16
                    const float* __restrict__ enorm,         // [K]
                    int* __restrict__ idx,                   // [N]
                    int* __restrict__ hist) {                // [K]
    __shared__ unsigned short ltile[16 * LDS_ROW];

    const int t    = threadIdx.x;
    const int lane = t & 31;
    const int wave = t >> 5;
    const int half = lane >> 4;     // 0: lanes 0-15, 1: lanes 16-31
    const int lrow = lane & 15;     // column (k) slot / A-row within half

    const int n0  = blockIdx.x * 128;     // block rows (never crosses batch)
    const int wn0 = n0 + wave * 16;       // this wave's first row
    const int b   = wn0 >> 10;
    const int hw  = (wn0 & 1023) + lrow;  // this lane's row position in HW

    // ---- build resident A fragments: x rows wn0..wn0+15, all 256 d, bf16 ----
    // A 16x32 bf16 layout: lane<16 row M=lane, K pairs {0..7,16..23};
    //                      lane>=16 row M=lane-16, K pairs {8..15,24..31}
    const float* xrow = inputs + (size_t)b * VQ_D * VQ_HW + hw;
    Frag16 a[8];
#pragma unroll
    for (int c = 0; c < 8; ++c) {
        const int dbase = c * 32 + (half ? 8 : 0);
#pragma unroll
        for (int j = 0; j < 8; ++j) {
            const int d = dbase + ((j < 4) ? (2 * j) : (16 + 2 * (j - 4)));
            float f0 = xrow[(size_t)d * VQ_HW];
            float f1 = xrow[(size_t)(d + 1) * VQ_HW];
            a[c].u[j] = (unsigned int)f32_to_bf16_rne(f0) |
                        ((unsigned int)f32_to_bf16_rne(f1) << 16);
        }
    }

    float best_d[8];
    int   best_k[8];
#pragma unroll
    for (int j = 0; j < 8; ++j) { best_d[j] = 3.4e38f; best_k[j] = 0; }

    // cooperative LDS staging coordinates: 16 rows x (128 u32), 16B chunks
    const int ldr = t >> 4;   // 0..15 : code row within tile
    const int ldc = t & 15;   // 0..15 : 32B chunk within row

#pragma unroll 1
    for (int kt = 0; kt < VQ_K; kt += 16) {
        // issue global loads for this tile before the barrier (overlap)
        const uint4* src =
            (const uint4*)(e_bf + (size_t)(kt + ldr) * VQ_D) + ldc * 2;
        uint4 v0 = src[0];
        uint4 v1 = src[1];
        if (kt + 16 < VQ_K)   // prefetch next tile -> global_prefetch_b8
            __builtin_prefetch(e_bf + (size_t)(kt + 16 + ldr) * VQ_D + ldc * 16, 0, 1);

        __syncthreads();      // previous tile fully consumed
        uint4* dst = (uint4*)(&ltile[ldr * LDS_ROW]) + ldc * 2;
        dst[0] = v0;
        dst[1] = v1;
        __syncthreads();      // tile ready

        // ---- 8 WMMAs over d, fp32 accumulate ----
        v8f acc = {};
        const unsigned short* brow = &ltile[lrow * LDS_ROW];
#pragma unroll
        for (int c = 0; c < 8; ++c) {
            Frag16 bf;
            const int dbase = c * 32 + (half ? 8 : 0);
#pragma unroll
            for (int j = 0; j < 8; ++j) {
                const int d = dbase + ((j < 4) ? (2 * j) : (16 + 2 * (j - 4)));
                bf.u[j] = *(const unsigned int*)(brow + d);  // ds_load_b32
            }
            acc = __builtin_amdgcn_wmma_f32_16x16x32_bf16(
                false, a[c].v, false, bf.v, (short)0, acc, false, false);
        }

        // C layout: VGPR j, lane -> (M = j + 8*half, Ncol = lrow)
        const int kk = kt + lrow;
        const float en = enorm[kk];
#pragma unroll
        for (int j = 0; j < 8; ++j) {
            float dist = __builtin_fmaf(-2.0f, acc[j], en);
            if (dist < best_d[j]) { best_d[j] = dist; best_k[j] = kk; }
        }
    }

    // ---- cross-lane argmin over the 16 k-columns of each output row ----
#pragma unroll
    for (int j = 0; j < 8; ++j) {
        float bd = best_d[j];
        int   bk = best_k[j];
#pragma unroll
        for (int m = 1; m < 16; m <<= 1) {
            float od = __shfl_xor(bd, m, 16);
            int   ok = __shfl_xor(bk, m, 16);
            if (od < bd || (od == bd && ok < bk)) { bd = od; bk = ok; }
        }
        if (lrow == 0) {
            const int row = wn0 + j + 8 * half;
            idx[row] = bk;
            atomicAdd(&hist[bk], 1);
        }
    }
}

// ---------------------------------------------------------------------------
// Kernel 2: gather quantized output + accumulate sum((q-x)^2)
// grid: B*D blocks x 256 threads, each thread 4 hw positions (coalesced)
// ---------------------------------------------------------------------------
__global__ void vq_out_loss(const float* __restrict__ inputs,
                            const float* __restrict__ emb,
                            const int* __restrict__ idx,
                            float* __restrict__ out,     // d_out + OUT_OFF
                            float* __restrict__ sumsq) {
    const int bd = blockIdx.x;           // b*256 + d
    const int b  = bd >> 8;
    const int d  = bd & 255;
    const int t  = threadIdx.x;

    const float* xs = inputs + (size_t)bd * VQ_HW;
    float*       os = out    + (size_t)bd * VQ_HW;
    const int*   ir = idx    + b * VQ_HW;

    float s = 0.0f;
#pragma unroll
    for (int i = 0; i < 4; ++i) {
        const int hw = t + i * 256;
        const int q  = ir[hw];
        const float e = emb[(size_t)q * VQ_D + d];
        const float x = xs[hw];
        os[hw] = e;                       // straight-through output == q
        const float df = e - x;
        s += df * df;
    }

    __shared__ float red[256];
    red[t] = s;
    __syncthreads();
    for (int st = 128; st > 0; st >>= 1) {
        if (t < st) red[t] += red[t + st];
        __syncthreads();
    }
    if (t == 0) atomicAdd(sumsq, red[0]);
}

// ---------------------------------------------------------------------------
// Kernel 3: one-hot encodings [N][K], b64 stores (ENC_OFF is 8B aligned)
// grid: N blocks x 256 threads, each thread 4 consecutive k
// ---------------------------------------------------------------------------
__global__ void vq_encodings(const int* __restrict__ idx,
                             float* __restrict__ enc) {   // d_out + ENC_OFF
    const int n = blockIdx.x;
    const int t = threadIdx.x;
    const int q = idx[n];
    float* row = enc + (size_t)n * VQ_K;
    const int k0 = t * 4;
    float2 v0, v1;
    v0.x = (k0     == q) ? 1.0f : 0.0f;
    v0.y = (k0 + 1 == q) ? 1.0f : 0.0f;
    v1.x = (k0 + 2 == q) ? 1.0f : 0.0f;
    v1.y = (k0 + 3 == q) ? 1.0f : 0.0f;
    *(float2*)(row + k0)     = v0;
    *(float2*)(row + k0 + 2) = v1;
}

// ---------------------------------------------------------------------------
// Kernel 4: scalars — loss and perplexity
// grid: 1 block x 256 threads
// ---------------------------------------------------------------------------
__global__ void vq_final(const int* __restrict__ hist,
                         const float* __restrict__ sumsq,
                         float* __restrict__ d_out) {
    const int t = threadIdx.x;
    float s = 0.0f;
#pragma unroll
    for (int i = 0; i < 4; ++i) {
        const int k = t + i * 256;
        const float p = (float)hist[k] * (1.0f / (float)VQ_N);
        s += p * logf(p + 1e-10f);
    }
    __shared__ float red[256];
    red[t] = s;
    __syncthreads();
    for (int st = 128; st > 0; st >>= 1) {
        if (t < st) red[t] += red[t + st];
        __syncthreads();
    }
    if (t == 0) {
        d_out[PERP_OFF] = expf(-red[0]);
        // loss = q_latent + beta * e_latent, both == mean((q-x)^2)
        d_out[0] = (1.0f + BETA_F) * (*sumsq) * (1.0f / ((float)VQ_N * (float)VQ_D));
    }
}

// ---------------------------------------------------------------------------
// launch
// ---------------------------------------------------------------------------
extern "C" void kernel_launch(void* const* d_in, const int* in_sizes, int n_in,
                              void* d_out, int out_size, void* d_ws, size_t ws_size,
                              hipStream_t stream) {
    const float* inputs = (const float*)d_in[0];   // [64,256,32,32]
    const float* emb    = (const float*)d_in[1];   // [1024,256]
    float* out = (float*)d_out;

    // workspace layout (bytes)
    char* ws = (char*)d_ws;
    unsigned short* e_bf  = (unsigned short*)(ws);                 // 524288 B
    float*          enorm = (float*)(ws + 524288);                 //   4096 B
    int*            idx   = (int*)  (ws + 528384);                 // 262144 B
    int*            hist  = (int*)  (ws + 790528);                 //   4096 B
    float*          sumsq = (float*)(ws + 794624);                 //      4 B

    vq_prep<<<VQ_K, VQ_D, 0, stream>>>(emb, e_bf, enorm, hist, sumsq);

    vq_dist_argmin<<<VQ_N / 128, 256, 0, stream>>>(inputs, e_bf, enorm, idx, hist);

    vq_out_loss<<<VQ_B * VQ_D, 256, 0, stream>>>(inputs, emb, idx,
                                                 out + OUT_OFF, sumsq);

    vq_encodings<<<VQ_N, 256, 0, stream>>>(idx, out + ENC_OFF);

    vq_final<<<1, 256, 0, stream>>>(hist, sumsq, out);
}
